// LocalAttention_6004364280646
// MI455X (gfx1250) — compile-verified, matrix operand order checked by reading, not run on hardware
//
#include <hip/hip_runtime.h>

// ---------------------------------------------------------------------------
// Local windowed attention, fused per 7x7 window, bf16 WMMA on gfx1250.
// B=16, C=256, X=Y=112, P=7 -> 4096 windows, n=49 tokens (padded to 64),
// HEADS=8, DIM_HEAD=64, INNER=512, SCALE=1/8.
//
// All LDS B-operand buffers use "fragment order": [kchunk][ntile][lane][8dw],
// so a WMMA B fragment is exactly two contiguous ds_load_b128 per lane and
// lands in a contiguous even-aligned VGPR tuple (no repacking moves).
// ---------------------------------------------------------------------------

typedef __attribute__((ext_vector_type(16))) __bf16      v16bf;
typedef __attribute__((ext_vector_type(8)))  float       v8f;
typedef __attribute__((ext_vector_type(4)))  unsigned int v4u;

union Frag {
    v16bf        v;
    unsigned int u[8];
    v4u          q[2];
};

__device__ __forceinline__ unsigned short f2bf(float x) {
    unsigned int u = __float_as_uint(x);
    unsigned int r = u + 0x7fffu + ((u >> 16) & 1u);   // round-to-nearest-even
    return (unsigned short)(r >> 16);
}

// K index pattern for 16-bit A/B fragments (16x32), per CDNA5 ISA 7.12.2:
// VGPR v<4: K=2v,2v+1 (lanes 0-15) / +8 (lanes 16-31); v>=4: K=2v+8 / +8.
__device__ __forceinline__ int frag_k0(int v, int half) {
    return ((v < 4) ? 2 * v : 2 * v + 8) + (half ? 8 : 0);
}

// Fragment-order element offset: byte offset of element (k, n) inside a
// fragment-order B buffer with 4 ntiles (64 columns).
__device__ __forceinline__ int bfrag_off(int k, int n) {
    int kchunk = k >> 5;
    int kk     = k & 31;
    int hb     = (kk >> 3) & 1;                    // lane half
    int u      = kk & ~8;                          // 0..7 or 16..23 (+parity)
    int v      = ((u >> 1) & 3) | ((u >> 2) & 4);  // VGPR index 0..7
    int lane   = hb * 16 + (n & 15);
    int ntile  = n >> 4;
    return (((kchunk * 4 + ntile) * 32 + lane) * 8 + v) * 4 + (kk & 1) * 2;
}

// B fragment: two contiguous b128 LDS loads.
__device__ __forceinline__ void load_bfrag_f(Frag& f, const char* base,
                                             int kchunk, int ntile, int lane) {
    const v4u* p = (const v4u*)(base + ((kchunk * 4 + ntile) * 32 + lane) * 32);
    f.q[0] = p[0];
    f.q[1] = p[1];
}

// A fragment from row-major LDS [row][64 bf16] (pitch 128B): per lane the 8
// dwords form two contiguous 16B runs -> two b128 LDS loads.
__device__ __forceinline__ void load_afrag_row(Frag& f, const char* rowbase,
                                               int kchunk, int half) {
    f.q[0] = *(const v4u*)(rowbase + kchunk * 64 + half * 16);
    f.q[1] = *(const v4u*)(rowbase + kchunk * 64 + 32 + half * 16);
}

__device__ __forceinline__ v8f wmma_bf16(const Frag& a, const Frag& b, v8f c) {
    return __builtin_amdgcn_wmma_f32_16x16x32_bf16(
        /*neg_a=*/false, a.v, /*neg_b=*/false, b.v,
        /*c_mod=*/(short)0, c, /*reuse_a=*/false, /*reuse_b=*/false);
}

// ---------------------------------------------------------------------------
// Kernel 1: pack Wq/Wkv (1536x256) and Wo (256x512) into bf16 A-fragment
// order: frag[mtile][kchunk][lane][16 bf16]  (32 bytes/lane, b128-loadable).
// ---------------------------------------------------------------------------
__global__ void pack_weights(const float* __restrict__ Wq,
                             const float* __restrict__ Wkv,
                             const float* __restrict__ Wo,
                             unsigned short* __restrict__ wqkv_pack,
                             unsigned short* __restrict__ wo_pack) {
    const int NQKV = 96 * 8 * 32 * 16;   // 393216 elements
    const int NWO  = 16 * 16 * 32 * 16;  // 131072 elements
    int idx = blockIdx.x * 256 + threadIdx.x;
    if (idx < NQKV) {
        int e = idx & 15, lane = (idx >> 4) & 31;
        int kc = (idx >> 9) & 7, mt = idx >> 12;
        int k   = kc * 32 + frag_k0(e >> 1, lane >> 4) + (e & 1);
        int row = mt * 16 + (lane & 15);               // 0..1535 combined
        float val = (row < 512) ? Wq[row * 256 + k]
                                : Wkv[(row - 512) * 256 + k];
        wqkv_pack[idx] = f2bf(val);
    } else {
        int i2 = idx - NQKV;
        if (i2 < NWO) {
            int e = i2 & 15, lane = (i2 >> 4) & 31;
            int kc = (i2 >> 9) & 15, mt = i2 >> 13;
            int k   = kc * 32 + frag_k0(e >> 1, lane >> 4) + (e & 1);
            int row = mt * 16 + (lane & 15);           // 0..255
            wo_pack[i2] = f2bf(Wo[row * 512 + k]);
        }
    }
}

// ---------------------------------------------------------------------------
// Kernel 2: fused window attention. grid = 4096 blocks, 256 threads (8 waves).
// LDS (dynamic, 224 KB):
//   fbuf  [0,      32768)  f window, fragment order (K=c 8 chunks, N=tok)
//   qbuf  [32768,  98304)  Q row-major [head][tok][d]  (A operand / probs)
//   kbuf  [98304, 163840)  K fragment order per head   (K=d,   N=tok)
//   vbuf  [163840,229376)  V fragment order per head (K=tok, N=d); reused as
//                          obuf fragment order (K=inner 16 chunks, N=tok).
// 224 KB LDS -> one block per WGP (2 waves/SIMD); register-resident B tiles
// are affordable (2 x ~450 VGPR < 1024/SIMD).
// ---------------------------------------------------------------------------
__global__ __launch_bounds__(256)
void local_attention_fused(const float* __restrict__ fmap,
                           const unsigned short* __restrict__ wqkv_pack,
                           const unsigned short* __restrict__ wo_pack,
                           const float* __restrict__ bo,
                           float* __restrict__ out) {
    extern __shared__ char smem[];
    char* fbuf = smem;
    char* qb   = smem + 32768;
    char* kb   = smem + 98304;
    char* vb   = smem + 163840;

    const int tid  = threadIdx.x;
    const int lane = tid & 31;
    const int wave = tid >> 5;
    const int half = lane >> 4;
    const int nl   = lane & 15;

    const int w  = blockIdx.x;          // window id
    const int bb = w >> 8;              // batch
    const int wx = (w >> 4) & 15;
    const int wy = w & 15;
    const long fbase = (long)bb * 256 * 12544 + (wx * 7) * 112 + (wy * 7);

    // ---- stage 0: window of fmap -> fbuf (bf16, zero-padded tokens 49..63)
    for (int idx = tid; idx < 256 * 64; idx += 256) {
        int c = idx >> 6, t = idx & 63;
        float val = 0.0f;
        if (t < 49) {
            int pi = t / 7, pj = t % 7;
            val = fmap[fbase + (long)c * 12544 + pi * 112 + pj];
        }
        *(unsigned short*)(fbuf + bfrag_off(c, t)) = f2bf(val);
    }
    __syncthreads();

    // ---- stage 1: QKV = W(1536x256) x f(256x64). 96 M-tiles, 12 per wave.
    for (int mt = wave * 12; mt < wave * 12 + 12; ++mt) {
        // prefetch next M-tile's A fragments toward the WGP (L2-resident)
        {
            const unsigned short* nxt =
                wqkv_pack + (((mt + 1) * 8) * 32 + lane) * 16;
#pragma unroll
            for (int kc = 0; kc < 8; ++kc)
                __builtin_prefetch((const void*)(nxt + kc * 512), 0, 0);
        }
        // all 8 A fragments up front: one clause of 16 b128 loads, waits
        // amortize across the 32 WMMAs below.
        Frag af[8];
#pragma unroll
        for (int kc = 0; kc < 8; ++kc) {
            const v4u* ap =
                (const v4u*)(wqkv_pack + ((mt * 8 + kc) * 32 + lane) * 16);
            af[kc].q[0] = ap[0];
            af[kc].q[1] = ap[1];
        }

        v8f acc[4];
#pragma unroll
        for (int n = 0; n < 4; ++n)
#pragma unroll
            for (int i = 0; i < 8; ++i) acc[n][i] = 0.0f;

#pragma unroll
        for (int kc = 0; kc < 8; ++kc) {
#pragma unroll
            for (int n = 0; n < 4; ++n) {
                Frag bf_;
                load_bfrag_f(bf_, fbuf, kc, n, lane);
                acc[n] = wmma_bf16(af[kc], bf_, acc[n]);
            }
        }

        // scatter C tiles: rows 0..511 of q / k / v
        const int kind = mt >> 5;            // 0=q, 1=k, 2=v
        const int mloc = mt & 31;
        const int row0 = mloc * 16 + half * 8;   // multiple of 8
        const int h    = row0 >> 6;
        const int d0   = row0 & 63;              // multiple of 8
#pragma unroll
        for (int n = 0; n < 4; ++n) {
            int tok = n * 16 + nl;
            if (kind == 0) {        // Q, pre-scaled, row-major [tok][d]
#pragma unroll
                for (int rr = 0; rr < 4; ++rr) {
                    unsigned int pk =
                        (unsigned int)f2bf(acc[n][2 * rr] * 0.125f) |
                        ((unsigned int)f2bf(acc[n][2 * rr + 1] * 0.125f) << 16);
                    *(unsigned int*)(qb + h * 8192 + tok * 128 +
                                     (d0 + 2 * rr) * 2) = pk;
                }
            } else if (kind == 1) { // K fragment order (K=d, N=tok)
#pragma unroll
                for (int rr = 0; rr < 4; ++rr) {
                    unsigned int pk =
                        (unsigned int)f2bf(acc[n][2 * rr]) |
                        ((unsigned int)f2bf(acc[n][2 * rr + 1]) << 16);
                    int d = d0 + 2 * rr;        // even -> dword aligned
                    *(unsigned int*)(kb + h * 8192 + bfrag_off(d, tok)) = pk;
                }
            } else {                // V fragment order (K=tok, N=d)
#pragma unroll
                for (int r = 0; r < 8; ++r) {
                    int d = d0 + r;
                    *(unsigned short*)(vb + h * 8192 + bfrag_off(tok, d)) =
                        f2bf(acc[n][r]);
                }
            }
        }
    }
    __syncthreads();

    // ---- stage 2: attention, one head per wave
    {
        const int h = wave;
        char* qh = qb + h * 8192;       // Q rows; reused in-place for probs
        char* kh = kb + h * 8192;
        char* vh = vb + h * 8192;

        Frag kf[4][2], vf[4][2];
#pragma unroll
        for (int n = 0; n < 4; ++n)
#pragma unroll
            for (int kc = 0; kc < 2; ++kc) {
                load_bfrag_f(kf[n][kc], kh, kc, n, lane);
                load_bfrag_f(vf[n][kc], vh, kc, n, lane);
            }
        asm volatile("" ::: "memory");  // pin K/V loads before obuf aliasing

        for (int mt = 0; mt < 4; ++mt) {
            const char* arow = qh + (mt * 16 + nl) * 128;
            Frag qf[2];
#pragma unroll
            for (int kc = 0; kc < 2; ++kc) load_afrag_row(qf[kc], arow, kc, half);

            v8f dots[4];
#pragma unroll
            for (int n = 0; n < 4; ++n) {
#pragma unroll
                for (int i = 0; i < 8; ++i) dots[n][i] = 0.0f;
#pragma unroll
                for (int kc = 0; kc < 2; ++kc)
                    dots[n] = wmma_bf16(qf[kc], kf[n][kc], dots[n]);
            }
            // mask padded columns, softmax per row (rows across 16 lanes)
#pragma unroll
            for (int n = 0; n < 4; ++n) {
                int j = n * 16 + nl;
#pragma unroll
                for (int r = 0; r < 8; ++r)
                    if (j >= 49) dots[n][r] = -1e30f;
            }
#pragma unroll
            for (int r = 0; r < 8; ++r) {
                float m = fmaxf(fmaxf(dots[0][r], dots[1][r]),
                                fmaxf(dots[2][r], dots[3][r]));
#pragma unroll
                for (int off = 1; off < 16; off <<= 1)
                    m = fmaxf(m, __shfl_xor(m, off, 16));
                float s = 0.0f;
#pragma unroll
                for (int n = 0; n < 4; ++n) {
                    dots[n][r] = __expf(dots[n][r] - m);
                    s += dots[n][r];
                }
#pragma unroll
                for (int off = 1; off < 16; off <<= 1)
                    s += __shfl_xor(s, off, 16);
                float inv = 1.0f / s;
#pragma unroll
                for (int n = 0; n < 4; ++n) dots[n][r] *= inv;
            }
            // probs -> LDS (overwrite this mtile's Q rows), then A-fragments
#pragma unroll
            for (int n = 0; n < 4; ++n) {
                int j = n * 16 + nl;
#pragma unroll
                for (int r = 0; r < 8; ++r) {
                    int i = mt * 16 + r + half * 8;
                    *(unsigned short*)(qh + i * 128 + j * 2) = f2bf(dots[n][r]);
                }
            }
            asm volatile("" ::: "memory");  // order LDS store -> load (in-wave)

            Frag pf[2];
#pragma unroll
            for (int kc = 0; kc < 2; ++kc) load_afrag_row(pf[kc], arow, kc, half);

            v8f oacc[4];
#pragma unroll
            for (int n = 0; n < 4; ++n) {
#pragma unroll
                for (int i = 0; i < 8; ++i) oacc[n][i] = 0.0f;
#pragma unroll
                for (int kc = 0; kc < 2; ++kc)
                    oacc[n] = wmma_bf16(pf[kc], vf[n][kc], oacc[n]);
            }
            // out tile -> obuf (aliases vbuf): fragment order (K=inner, N=tok)
#pragma unroll
            for (int n = 0; n < 4; ++n) {
                int d = n * 16 + nl;
#pragma unroll
                for (int r = 0; r < 8; ++r) {
                    int i = mt * 16 + r + half * 8;   // token
                    *(unsigned short*)(vb + bfrag_off(h * 64 + d, i)) =
                        f2bf(oacc[n][r]);
                }
            }
        }
    }
    __syncthreads();

    // ---- stage 3: out = Wo(256x512) x obuf(512x64) + bo; scatter to NCHW
    for (int mt = wave * 2; mt < wave * 2 + 2; ++mt) {
        v8f acc[4];
#pragma unroll
        for (int n = 0; n < 4; ++n)
#pragma unroll
            for (int i = 0; i < 8; ++i) acc[n][i] = 0.0f;

        // double-buffered groups of 4 A fragments (16 k-chunks total)
        Frag afg[2][4];
#pragma unroll
        for (int kk = 0; kk < 4; ++kk) {
            const v4u* ap =
                (const v4u*)(wo_pack + ((mt * 16 + kk) * 32 + lane) * 16);
            afg[0][kk].q[0] = ap[0];
            afg[0][kk].q[1] = ap[1];
        }
#pragma unroll
        for (int g = 0; g < 4; ++g) {
            const int cur = g & 1;
            if (g < 3) {
#pragma unroll
                for (int kk = 0; kk < 4; ++kk) {
                    const v4u* ap = (const v4u*)(wo_pack +
                        ((mt * 16 + (g + 1) * 4 + kk) * 32 + lane) * 16);
                    afg[cur ^ 1][kk].q[0] = ap[0];
                    afg[cur ^ 1][kk].q[1] = ap[1];
                }
            }
#pragma unroll
            for (int kk = 0; kk < 4; ++kk) {
                const int kc = g * 4 + kk;
#pragma unroll
                for (int n = 0; n < 4; ++n) {
                    Frag bf_;
                    load_bfrag_f(bf_, vb, kc, n, lane);
                    acc[n] = wmma_bf16(afg[cur][kk], bf_, acc[n]);
                }
            }
        }
#pragma unroll
        for (int n = 0; n < 4; ++n) {
            int tok = n * 16 + nl;
            if (tok < 49) {
                int pi = tok / 7, pj = tok % 7;
#pragma unroll
                for (int r = 0; r < 8; ++r) {
                    int o = mt * 16 + r + half * 8;   // 0..255
                    out[((long)(bb * 256 + o)) * 12544 +
                        (wx * 7 + pi) * 112 + (wy * 7 + pj)] = acc[n][r] + bo[o];
                }
            }
        }
    }
}

// ---------------------------------------------------------------------------
extern "C" void kernel_launch(void* const* d_in, const int* in_sizes, int n_in,
                              void* d_out, int out_size, void* d_ws, size_t ws_size,
                              hipStream_t stream) {
    const float* fmap = (const float*)d_in[0];
    const float* Wq   = (const float*)d_in[1];
    const float* Wkv  = (const float*)d_in[2];
    const float* Wo   = (const float*)d_in[3];
    const float* bo   = (const float*)d_in[4];
    float*       out  = (float*)d_out;

    unsigned short* wqkv_pack = (unsigned short*)d_ws;            // 768 KB
    unsigned short* wo_pack   = wqkv_pack + 96 * 8 * 32 * 16;     // 256 KB

    pack_weights<<<2048, 256, 0, stream>>>(Wq, Wkv, Wo, wqkv_pack, wo_pack);

    const size_t lds_bytes = 229376;   // 224 KB of the 320 KB WGP budget
    local_attention_fused<<<4096, 256, lds_bytes, stream>>>(
        fmap, wqkv_pack, wo_pack, bo, out);
}
